// Hypernetwork_86217173500424
// MI455X (gfx1250) — compile-verified, matrix operand order checked by reading, not run on hardware
//
#include <hip/hip_runtime.h>
#include <hip/hip_bf16.h>

typedef _Float16 h16 __attribute__((ext_vector_type(16)));
typedef _Float16 h8  __attribute__((ext_vector_type(8)));
typedef float    f8  __attribute__((ext_vector_type(8)));

#define B_      2
#define HW      96
#define NPIX    18432      // 2*96*96
#define P_TOT   294912     // 2*384*384
#define C_ENC   256
#define C_MID   512
#define NOUT    595
#define NT_OUT  38         // 608/16
#define KC_OUT  16         // 512/32
#define KC_REF  72         // 2304/32
#define NT_REF  32         // 512/16

__device__ __forceinline__ int clampi(int v, int lo, int hi) {
    return v < lo ? lo : (v > hi ? hi : v);
}

// k-offset within a 32-wide K chunk for B/A fragment half j on lane `lane`
__device__ __forceinline__ int frag_koff(int lane, int j) {
    return ((lane >= 16) ? 8 : 0) + ((j >= 8) ? 16 : 0) + (j & 7);
}

// ---------------- prep: out_w [512,595] f32 -> B-fragments f16 [38][16][32][16]
__global__ void k_prep_outw(const float* __restrict__ out_w, _Float16* __restrict__ bf) {
    int idx = blockIdx.x * blockDim.x + threadIdx.x;
    if (idx >= NT_OUT * KC_OUT * 32 * 16) return;
    int j    = idx & 15;
    int lane = (idx >> 4) & 31;
    int kc   = (idx >> 9) & 15;
    int nt   = idx >> 13;
    int n = nt * 16 + (lane & 15);
    int k = kc * 32 + frag_koff(lane, j);
    bf[idx] = (n < NOUT) ? (_Float16)out_w[(size_t)k * NOUT + n] : (_Float16)0.0f;
}

// ---------------- prep: ref_w [3,3,256,512] f32 -> B-fragments f16 [32][72][32][16]
__global__ void k_prep_refw(const float* __restrict__ ref_w, _Float16* __restrict__ bf) {
    int idx = blockIdx.x * blockDim.x + threadIdx.x;
    if (idx >= NT_REF * KC_REF * 32 * 16) return;
    int j    = idx & 15;
    int lane = (idx >> 4) & 31;
    int ntkc = idx >> 9;
    int kc   = ntkc % KC_REF;
    int nt   = ntkc / KC_REF;
    int n = nt * 16 + (lane & 15);
    int k = kc * 32 + frag_koff(lane, j);          // k = tap*256 + ci
    bf[idx] = (_Float16)ref_w[(size_t)k * C_MID + n];
}

// ---------------- conv1: 3->256, 3x3 SAME, ReLU, f32 compute, f16 out
__global__ void k_conv1(const float* __restrict__ src, const float* __restrict__ w,
                        const float* __restrict__ bias, _Float16* __restrict__ enc) {
    int idx = blockIdx.x * blockDim.x + threadIdx.x;
    if (idx >= NPIX * C_ENC) return;
    int co = idx & 255;
    int p  = idx >> 8;
    int b  = p / (HW * HW);
    int yy = (p % (HW * HW)) / HW;
    int xx = p % HW;
    float acc = bias[co];
    #pragma unroll
    for (int ky = 0; ky < 3; ++ky) {
        int iy = yy + ky - 1;
        if (iy < 0 || iy >= HW) continue;
        #pragma unroll
        for (int kx = 0; kx < 3; ++kx) {
            int ix = xx + kx - 1;
            if (ix < 0 || ix >= HW) continue;
            const float* sp = src + ((size_t)(b * HW + iy) * HW + ix) * 3;
            const float* wp = w + (size_t)((ky * 3 + kx) * 3) * C_ENC + co;
            #pragma unroll
            for (int ci = 0; ci < 3; ++ci)
                acc = fmaf(sp[ci], wp[(size_t)ci * C_ENC], acc);
        }
    }
    enc[(size_t)p * C_ENC + co] = (_Float16)(acc > 0.0f ? acc : 0.0f);
}

// ---------------- conv2: implicit GEMM, M=18432 K=2304 N=512, WMMA f16->f32
// Double-buffered im2col staging via CDNA5 async global->LDS copies (ASYNCcnt):
// stage of chunk kc+1 overlaps WMMA compute of chunk kc; one barrier per chunk.
__global__ void __launch_bounds__(128)
k_conv2(const _Float16* __restrict__ enc, const _Float16* __restrict__ bfrag,
        const float* __restrict__ ref_b, float* __restrict__ encod) {
    __shared__ _Float16 sA[2][64 * 32];
    const int tid   = threadIdx.x;
    const int lane  = tid & 31;
    const int wv    = tid >> 5;
    const bool hi   = lane >= 16;
    const int l15   = lane & 15;
    const int mbase = blockIdx.x * 64;
    const int nblk  = blockIdx.y;   // 64 N cols

    f8 acc0 = {}, acc1 = {}, acc2 = {}, acc3 = {};

    // per-thread staging geometry
    const int srow  = tid >> 1;
    const int shalf = (tid & 1) * 16;           // halves
    const int sp_   = mbase + srow;
    const int sb    = sp_ / (HW * HW);
    const int syy   = (sp_ % (HW * HW)) / HW;
    const int sxx   = sp_ % HW;
    const unsigned ldsoff0 = (unsigned)(uintptr_t)(&sA[0][srow * 32 + shalf]);
    const unsigned ldsoff1 = (unsigned)(uintptr_t)(&sA[1][srow * 32 + shalf]);

    auto stage = [&](int kc, int buf) {
        int tap = kc >> 3;
        int ci0 = (kc & 7) * 32 + shalf;
        int iy  = syy + tap / 3 - 1;
        int ix  = sxx + tap % 3 - 1;
        unsigned lo = buf ? ldsoff1 : ldsoff0;
        if (iy >= 0 && iy < HW && ix >= 0 && ix < HW) {
            const _Float16* gp =
                enc + ((size_t)(sb * HW + iy) * HW + ix) * C_ENC + ci0;
            // CDNA5 async copy: 2 x 16B global -> LDS, tracked by ASYNCcnt
            asm volatile(
                "global_load_async_to_lds_b128 %0, %1, off\n\t"
                "global_load_async_to_lds_b128 %0, %1, off offset:16"
                :: "v"(lo), "v"(gp) : "memory");
        } else {
            h8 z = {};
            h8* dp = (h8*)&sA[buf][srow * 32 + shalf];
            dp[0] = z; dp[1] = z;
        }
    };

    stage(0, 0);   // prologue

    for (int kc = 0; kc < KC_REF; ++kc) {
        const int cur = kc & 1;
        // own-wave async stage of chunk kc complete, then make all waves'
        // contributions visible (barrier also retires DS reads of buf cur^1)
        asm volatile("s_wait_asynccnt 0x0" ::: "memory");
        __syncthreads();
        if (kc + 1 < KC_REF) stage(kc + 1, cur ^ 1);   // overlap with compute

        // A fragment for this wave's 16 rows
        int arow = wv * 16 + l15;
        int abase = arow * 32 + (hi ? 8 : 0);
        h8 lo = *(const h8*)&sA[cur][abase];
        h8 hv = *(const h8*)&sA[cur][abase + 16];
        h16 a;
        #pragma unroll
        for (int i = 0; i < 8; ++i) { a[i] = lo[i]; a[i + 8] = hv[i]; }

        #pragma unroll
        for (int nt = 0; nt < 4; ++nt) {
            int ntg = nblk * 4 + nt;
            const h16 bmat = *(const h16*)(bfrag + ((size_t)(ntg * KC_REF + kc) * 32 + lane) * 16);
            f8 acc = (nt == 0) ? acc0 : (nt == 1) ? acc1 : (nt == 2) ? acc2 : acc3;
            acc = __builtin_amdgcn_wmma_f32_16x16x32_f16(false, a, false, bmat,
                                                         (short)0, acc, false, false);
            if (nt == 0) acc0 = acc; else if (nt == 1) acc1 = acc;
            else if (nt == 2) acc2 = acc; else acc3 = acc;
        }
    }

    // store D tiles: f32 encoding [p][n]
    #pragma unroll
    for (int nt = 0; nt < 4; ++nt) {
        f8 acc = (nt == 0) ? acc0 : (nt == 1) ? acc1 : (nt == 2) ? acc2 : acc3;
        int n = nblk * 64 + nt * 16 + l15;
        float bb = ref_b[n];
        #pragma unroll
        for (int r = 0; r < 8; ++r) {
            int m = r + (hi ? 8 : 0);
            int p = mbase + wv * 16 + m;
            encod[(size_t)p * C_MID + n] = acc[r] + bb;
        }
    }
}

// ---------------- fused: bilinear sample -> GEMM (K=512, N=595 pad 608) -> segmented store
// Each wave register-blocks 4 M-subtiles per B fragment (4x less B traffic).
__global__ void __launch_bounds__(128)
k_interp_out(const float* __restrict__ coords, const float* __restrict__ encod,
             const _Float16* __restrict__ bfrag, const float* __restrict__ out_b,
             float* __restrict__ out) {
    __shared__ _Float16 sF[64 * C_MID];   // 64 KB
    const int tid   = threadIdx.x;
    const int lane  = tid & 31;
    const int wv    = tid >> 5;
    const bool hi   = lane >= 16;
    const int l15   = lane & 15;
    const int pbase = blockIdx.x * 64;

    // Phase 1: bilinear sample, one source row at a time.
    // All per-row parameters are block-uniform -> SALU/scalar loads; the inner
    // loop is 4 coalesced 512B gathers + 4 FMAs per element.
    for (int row = 0; row < 64; ++row) {
        int p = pbase + row;
        int b = p / (384 * 384);
        float cy = coords[(size_t)p * 2 + 0];
        float cx = coords[(size_t)p * 2 + 1];
        float y = cy * 96.0f + 47.5f;
        float x = cx * 96.0f + 47.5f;
        float y0f = floorf(y), x0f = floorf(x);
        float wy = y - y0f, wx = x - x0f;
        int y0 = clampi((int)y0f, 0, HW - 1);
        int y1 = clampi((int)y0f + 1, 0, HW - 1);
        int x0 = clampi((int)x0f, 0, HW - 1);
        int x1 = clampi((int)x0f + 1, 0, HW - 1);
        float w00 = (1.0f - wy) * (1.0f - wx);
        float w01 = (1.0f - wy) * wx;
        float w10 = wy * (1.0f - wx);
        float w11 = wy * wx;
        size_t rb = (size_t)b * HW;
        const float* e00 = encod + ((rb + y0) * HW + x0) * C_MID;
        const float* e01 = encod + ((rb + y0) * HW + x1) * C_MID;
        const float* e10 = encod + ((rb + y1) * HW + x0) * C_MID;
        const float* e11 = encod + ((rb + y1) * HW + x1) * C_MID;
        #pragma unroll
        for (int i = 0; i < 4; ++i) {
            int c = i * 128 + tid;
            float v = w00 * e00[c] + w01 * e01[c] + w10 * e10[c] + w11 * e11[c];
            sF[row * C_MID + c] = (_Float16)v;
        }
    }
    __syncthreads();

    // Phase 2: wave w sweeps N-tiles nt = w, w+4, ... ; for each it computes
    // all 4 M-subtiles (64 rows) reusing one B fragment per K-chunk.
    for (int nt = wv; nt < NT_OUT; nt += 4) {
        f8 acc0 = {}, acc1 = {}, acc2 = {}, acc3 = {};
        for (int kc = 0; kc < KC_OUT; ++kc) {
            const h16 bmat = *(const h16*)(bfrag + ((size_t)(nt * KC_OUT + kc) * 32 + lane) * 16);
            #pragma unroll
            for (int ms = 0; ms < 4; ++ms) {
                int arow  = ms * 16 + l15;
                int abase = arow * C_MID + kc * 32 + (hi ? 8 : 0);
                h8 lo = *(const h8*)&sF[abase];
                h8 hv = *(const h8*)&sF[abase + 16];
                h16 a;
                #pragma unroll
                for (int i = 0; i < 8; ++i) { a[i] = lo[i]; a[i + 8] = hv[i]; }
                f8 acc = (ms == 0) ? acc0 : (ms == 1) ? acc1 : (ms == 2) ? acc2 : acc3;
                acc = __builtin_amdgcn_wmma_f32_16x16x32_f16(false, a, false, bmat,
                                                             (short)0, acc, false, false);
                if (ms == 0) acc0 = acc; else if (ms == 1) acc1 = acc;
                else if (ms == 2) acc2 = acc; else acc3 = acc;
            }
        }
        // segmented store (tuple outputs concatenated flat)
        int c = nt * 16 + l15;
        if (c < NOUT) {
            long long segbase; int ss, s0;
            if      (c < 256) { segbase = 0LL;                 ss = 256; s0 = 0;   }
            else if (c < 272) { segbase = 294912LL * 256;      ss = 16;  s0 = 256; }
            else if (c < 528) { segbase = 294912LL * 272;      ss = 256; s0 = 272; }
            else if (c < 544) { segbase = 294912LL * 528;      ss = 16;  s0 = 528; }
            else if (c < 592) { segbase = 294912LL * 544;      ss = 48;  s0 = 544; }
            else              { segbase = 294912LL * 592;      ss = 3;   s0 = 592; }
            float bb = out_b[c];
            #pragma unroll
            for (int ms = 0; ms < 4; ++ms) {
                f8 acc = (ms == 0) ? acc0 : (ms == 1) ? acc1 : (ms == 2) ? acc2 : acc3;
                #pragma unroll
                for (int r = 0; r < 8; ++r) {
                    int m = r + (hi ? 8 : 0);
                    long long p = pbase + ms * 16 + m;
                    out[segbase + p * ss + (c - s0)] = acc[r] + bb;
                }
            }
        }
    }
}

extern "C" void kernel_launch(void* const* d_in, const int* in_sizes, int n_in,
                              void* d_out, int out_size, void* d_ws, size_t ws_size,
                              hipStream_t stream) {
    const float* src      = (const float*)d_in[0];
    const float* coords   = (const float*)d_in[1];
    const float* enc_w    = (const float*)d_in[2];
    const float* enc_b    = (const float*)d_in[3];
    const float* ref_w    = (const float*)d_in[4];
    const float* ref_b    = (const float*)d_in[5];
    const float* out_w    = (const float*)d_in[6];
    const float* out_b    = (const float*)d_in[7];
    float* out = (float*)d_out;

    char* ws = (char*)d_ws;
    _Float16* enc       = (_Float16*)(ws);                                   // 9,437,184 B
    float*    encod     = (float*)   (ws + 9437184);                         // 37,748,736 B
    _Float16* bfrag_out = (_Float16*)(ws + 9437184 + 37748736);              // 622,592 B
    _Float16* bfrag_ref = (_Float16*)(ws + 9437184 + 37748736 + 622592);     // 2,359,296 B

    k_prep_outw<<<(NT_OUT * KC_OUT * 32 * 16 + 255) / 256, 256, 0, stream>>>(out_w, bfrag_out);
    k_prep_refw<<<(NT_REF * KC_REF * 32 * 16 + 255) / 256, 256, 0, stream>>>(ref_w, bfrag_ref);
    k_conv1<<<(NPIX * C_ENC + 255) / 256, 256, 0, stream>>>(src, enc_w, enc_b, enc);
    k_conv2<<<dim3(NPIX / 64, C_MID / 64), 128, 0, stream>>>(enc, bfrag_ref, ref_b, encod);
    k_interp_out<<<P_TOT / 64, 128, 0, stream>>>(coords, encod, bfrag_out, out_b, out);
}
